// CausalFractalProcessor_37941741093001
// MI455X (gfx1250) — compile-verified, matrix operand order checked by reading, not run on hardware
//
#include <hip/hip_runtime.h>
#include <math.h>

// ---------------------------------------------------------------------------
// CausalFractalProcessor forward for MI455X (gfx1250, wave32, WMMA).
// - one bf16-WMMA tiled GEMM (f32 accumulate) for all matmuls
// - head GEMM (4096x32000x2048) uses bf16-resident operands staged to LDS
//   with global_load_async_to_lds_b128 (ASYNCcnt) and double buffering
// ---------------------------------------------------------------------------

typedef __attribute__((ext_vector_type(16))) __bf16        bf16x16;
typedef __attribute__((ext_vector_type(8)))  float         v8f;
typedef __attribute__((ext_vector_type(4)))  float         f32x4;
typedef __attribute__((ext_vector_type(4)))  unsigned short u16x4;

#define MTOK 4096   // B*L
#define DMODEL 512

__device__ __forceinline__ unsigned short f2bf(float f) {
    unsigned u = __float_as_uint(f);
    unsigned r = u + 0x7FFFu + ((u >> 16) & 1u);   // round-to-nearest-even
    return (unsigned short)(r >> 16);
}

__device__ __forceinline__ float gelu_exact(float v) {
    return 0.5f * v * (1.0f + erff(v * 0.70710678118654752f));
}

// ---- gfx1250 async copy: global -> LDS, 16B per lane, tracked by ASYNCcnt ----
__device__ __forceinline__ void async_load_b128(const void* gaddr, unsigned lds_off) {
#if defined(__HIP_DEVICE_COMPILE__) && defined(__gfx1250__)
    asm volatile("global_load_async_to_lds_b128 %0, %1, off"
                 :: "v"(lds_off), "v"(gaddr)
                 : "memory");
#else
    (void)gaddr; (void)lds_off;
#endif
}
__device__ __forceinline__ void wait_asynccnt0() {
#if defined(__HIP_DEVICE_COMPILE__) && defined(__gfx1250__)
    asm volatile("s_wait_asynccnt 0" ::: "memory");
#endif
}

// ---------------------------------------------------------------------------
// WMMA GEMM (f32 operands, converted on the fly): C = A[M,K] * B^T (+bias)(+GELU)
//   A: row-major, row stride lda (K contiguous)
//   B: bTrans==0: (N,K) row-major  |  bTrans==1: (K,N) row-major
//   C: f32 (OUT_BF16=0) or packed bf16 (OUT_BF16=1); row stride ldcv, col offset
// Tile 128x128x32, 256 threads = 8 waves (4 on M x 2 on N), 8 WMMA/wave/K-step.
// ---------------------------------------------------------------------------
#define BM 128
#define BN 128
#define BK 32
#define LDSP 40      // (BK + 8) ushorts per row -> 80B, 16B-aligned rows

template <int FUSE_GELU, int OUT_BF16>
__global__ __launch_bounds__(256)
void gemm_wmma_bf16(const float* __restrict__ A, int lda,
                    const float* __restrict__ B, int ldb, int bTrans,
                    const float* __restrict__ bias,
                    void* __restrict__ Cv, int ldcv, int ccol,
                    int M, int N, int K)
{
    __shared__ __align__(16) unsigned short As[2][BM][LDSP];
    __shared__ __align__(16) unsigned short Bs[2][BN][LDSP];

    const int tid  = threadIdx.x;
    const int lane = tid & 31;
    const int wave = tid >> 5;
    const int wm   = wave & 3;       // 4 waves over M (32 rows each)
    const int wn   = wave >> 2;      // 2 waves over N (64 cols each)
    const int bm   = blockIdx.y * BM;
    const int bn   = blockIdx.x * BN;

    v8f acc[2][4] = {};
    const int nk = K / BK;

    auto stage = [&](int kb, int buf) {
        const int k0 = kb * BK;
        int r  = tid >> 3;           // 0..31
        const int cb = tid & 7;      // 4-col block
        for (int it = 0; it < 4; ++it, r += 32) {
            {
                const float* ap = A + (size_t)(bm + r) * lda + k0 + cb * 4;
                __builtin_prefetch((const void*)(ap + BK), 0, 1);
                f32x4 v = *reinterpret_cast<const f32x4*>(ap);
                u16x4 pk;
                pk.x = f2bf(v.x); pk.y = f2bf(v.y);
                pk.z = f2bf(v.z); pk.w = f2bf(v.w);
                *reinterpret_cast<u16x4*>(&As[buf][r][cb * 4]) = pk;
            }
            if (!bTrans) {
                const float* bp = B + (size_t)(bn + r) * ldb + k0 + cb * 4;
                __builtin_prefetch((const void*)(bp + BK), 0, 1);
                f32x4 v = *reinterpret_cast<const f32x4*>(bp);
                u16x4 pk;
                pk.x = f2bf(v.x); pk.y = f2bf(v.y);
                pk.z = f2bf(v.z); pk.w = f2bf(v.w);
                *reinterpret_cast<u16x4*>(&Bs[buf][r][cb * 4]) = pk;
            } else {
                u16x4 pk;
                pk.x = f2bf(B[(size_t)(k0 + cb * 4 + 0) * ldb + bn + r]);
                pk.y = f2bf(B[(size_t)(k0 + cb * 4 + 1) * ldb + bn + r]);
                pk.z = f2bf(B[(size_t)(k0 + cb * 4 + 2) * ldb + bn + r]);
                pk.w = f2bf(B[(size_t)(k0 + cb * 4 + 3) * ldb + bn + r]);
                *reinterpret_cast<u16x4*>(&Bs[buf][r][cb * 4]) = pk;
            }
        }
    };

    stage(0, 0);

    for (int kb = 0; kb < nk; ++kb) {
        __syncthreads();
        if (kb + 1 < nk) stage(kb + 1, (kb + 1) & 1);

        const int buf  = kb & 1;
        const int row  = lane & 15;
        const int koff = (lane >> 4) * 16;

        bf16x16 afr[2], bfr[4];
#pragma unroll
        for (int mi = 0; mi < 2; ++mi)
            afr[mi] = *reinterpret_cast<const bf16x16*>(
                &As[buf][wm * 32 + mi * 16 + row][koff]);
#pragma unroll
        for (int ni = 0; ni < 4; ++ni)
            bfr[ni] = *reinterpret_cast<const bf16x16*>(
                &Bs[buf][wn * 64 + ni * 16 + row][koff]);

#pragma unroll
        for (int mi = 0; mi < 2; ++mi)
#pragma unroll
            for (int ni = 0; ni < 4; ++ni)
                acc[mi][ni] = __builtin_amdgcn_wmma_f32_16x16x32_bf16(
                    false, afr[mi], false, bfr[ni],
                    (short)0, acc[mi][ni], false, false);
    }

    // C/D layout: VGPR e, lane c -> M = e + 8*(c>>4), N = c&15
    const int crow  = (lane >> 4) * 8;
    const int ccolL = lane & 15;
#pragma unroll
    for (int mi = 0; mi < 2; ++mi) {
#pragma unroll
        for (int ni = 0; ni < 4; ++ni) {
            const int gc = bn + wn * 64 + ni * 16 + ccolL;
            const float bv = bias ? bias[gc] : 0.0f;
#pragma unroll
            for (int e = 0; e < 8; ++e) {
                const int gr = bm + wm * 32 + mi * 16 + crow + e;
                float v = acc[mi][ni][e] + bv;
                if (FUSE_GELU) v = gelu_exact(v);
                if (gr < M && gc < N) {
                    if (OUT_BF16)
                        ((unsigned short*)Cv)[(size_t)gr * ldcv + ccol + gc] = f2bf(v);
                    else
                        ((float*)Cv)[(size_t)gr * ldcv + ccol + gc] = v;
                }
            }
        }
    }
}

// ---------------------------------------------------------------------------
// WMMA GEMM, bf16-resident operands, async-staged LDS tiles.
//   A: (M,K) bf16 row-major, stride lda; B: (N,K) bf16 row-major, stride ldb
//   C: f32, stride ldcv. Used for the vocab-projection head.
// ---------------------------------------------------------------------------
__global__ __launch_bounds__(256)
void gemm_wmma_bf16_async(const unsigned short* __restrict__ A, int lda,
                          const unsigned short* __restrict__ B, int ldb,
                          const float* __restrict__ bias,
                          float* __restrict__ C, int ldcv,
                          int M, int N, int K)
{
    __shared__ __align__(16) unsigned short As[2][BM][LDSP];
    __shared__ __align__(16) unsigned short Bs[2][BN][LDSP];

    const int tid  = threadIdx.x;
    const int lane = tid & 31;
    const int wave = tid >> 5;
    const int wm   = wave & 3;
    const int wn   = wave >> 2;
    const int bm   = blockIdx.y * BM;
    const int bn   = blockIdx.x * BN;

    v8f acc[2][4] = {};
    const int nk = K / BK;

    // 128 rows x 64B per tile = 512 x 16B segments; 2 segments/thread/matrix.
    auto stage = [&](int kb, int buf) {
        const int k0 = kb * BK;
#pragma unroll
        for (int it = 0; it < 2; ++it) {
            const int seg = it * 256 + tid;
            const int r = seg >> 2;          // row 0..127
            const int s = seg & 3;           // 16B segment in row
            const unsigned short* ga = A + (size_t)(bm + r) * lda + k0 + s * 8;
            async_load_b128((const void*)ga,
                            (unsigned)(size_t)&As[buf][r][s * 8]);
            const unsigned short* gb = B + (size_t)(bn + r) * ldb + k0 + s * 8;
            async_load_b128((const void*)gb,
                            (unsigned)(size_t)&Bs[buf][r][s * 8]);
        }
    };

    stage(0, 0);

    for (int kb = 0; kb < nk; ++kb) {
        wait_asynccnt0();        // this wave's async loads have landed in LDS
        __syncthreads();         // everyone's loads landed; prev readers done
        if (kb + 1 < nk) stage(kb + 1, (kb + 1) & 1);

        const int buf  = kb & 1;
        const int row  = lane & 15;
        const int koff = (lane >> 4) * 16;

        bf16x16 afr[2], bfr[4];
#pragma unroll
        for (int mi = 0; mi < 2; ++mi)
            afr[mi] = *reinterpret_cast<const bf16x16*>(
                &As[buf][wm * 32 + mi * 16 + row][koff]);
#pragma unroll
        for (int ni = 0; ni < 4; ++ni)
            bfr[ni] = *reinterpret_cast<const bf16x16*>(
                &Bs[buf][wn * 64 + ni * 16 + row][koff]);

#pragma unroll
        for (int mi = 0; mi < 2; ++mi)
#pragma unroll
            for (int ni = 0; ni < 4; ++ni)
                acc[mi][ni] = __builtin_amdgcn_wmma_f32_16x16x32_bf16(
                    false, afr[mi], false, bfr[ni],
                    (short)0, acc[mi][ni], false, false);
    }

    const int crow  = (lane >> 4) * 8;
    const int ccolL = lane & 15;
#pragma unroll
    for (int mi = 0; mi < 2; ++mi) {
#pragma unroll
        for (int ni = 0; ni < 4; ++ni) {
            const int gc = bn + wn * 64 + ni * 16 + ccolL;
            const float bv = bias ? bias[gc] : 0.0f;
#pragma unroll
            for (int e = 0; e < 8; ++e) {
                const int gr = bm + wm * 32 + mi * 16 + crow + e;
                if (gr < M && gc < N)
                    C[(size_t)gr * ldcv + gc] = acc[mi][ni][e] + bv;
            }
        }
    }
}

// ---------------------------------------------------------------------------
// Elementwise / small kernels
// ---------------------------------------------------------------------------
__global__ void embed_kernel(const int* __restrict__ tok,
                             const float* __restrict__ emb,
                             float* __restrict__ out, int n)
{
    int idx = blockIdx.x * blockDim.x + threadIdx.x;
    if (idx >= n) return;
    int row = idx >> 9, d = idx & 511;
    out[idx] = emb[(size_t)tok[row] * DMODEL + d];
}

__global__ void f32_to_bf16_kernel(const float* __restrict__ src,
                                   unsigned short* __restrict__ dst, long n4)
{
    long i = ((long)blockIdx.x * blockDim.x + threadIdx.x) * 4;
    if (i + 3 >= n4 * 4 + 3) return;   // n4 = n/4 vectors
    if (i >= n4 * 4) return;
    f32x4 v = *reinterpret_cast<const f32x4*>(src + i);
    u16x4 p;
    p.x = f2bf(v.x); p.y = f2bf(v.y); p.z = f2bf(v.z); p.w = f2bf(v.w);
    *reinterpret_cast<u16x4*>(dst + i) = p;
}

__global__ __launch_bounds__(256)
void posln_kernel(const float* __restrict__ x, int lda,
                  const float* __restrict__ phases,
                  const float* __restrict__ g, const float* __restrict__ b,
                  float* __restrict__ out, int L)
{
    __shared__ float red[256];
    const int row = blockIdx.x;
    const float l = (float)(row % L);
    const int t = threadIdx.x;
    const int d0 = t, d1 = t + 256;

    float p0 = l * phases[d0], p1 = l * phases[d1];
    float v0 = x[(size_t)row * lda + d0] * (1.0f + sinf(p0)) + cosf(p0);
    float v1 = x[(size_t)row * lda + d1] * (1.0f + sinf(p1)) + cosf(p1);

    red[t] = v0 + v1;
    __syncthreads();
    for (int s = 128; s > 0; s >>= 1) {
        if (t < s) red[t] += red[t + s];
        __syncthreads();
    }
    const float mu = red[0] * (1.0f / 512.0f);
    __syncthreads();
    const float c0 = v0 - mu, c1 = v1 - mu;
    red[t] = c0 * c0 + c1 * c1;
    __syncthreads();
    for (int s = 128; s > 0; s >>= 1) {
        if (t < s) red[t] += red[t + s];
        __syncthreads();
    }
    const float rstd = rsqrtf(red[0] * (1.0f / 512.0f) + 1e-5f);
    out[(size_t)row * DMODEL + d0] = c0 * rstd * g[d0] + b[d0];
    out[(size_t)row * DMODEL + d1] = c1 * rstd * g[d1] + b[d1];
}

__global__ void conv_kernel(const float* __restrict__ si,
                            const float* __restrict__ w0, const float* __restrict__ b0,
                            const float* __restrict__ w1, const float* __restrict__ b1,
                            const float* __restrict__ w2, const float* __restrict__ b2,
                            float* __restrict__ out, int L, int k, int dil, int n)
{
    int idx = blockIdx.x * blockDim.x + threadIdx.x;
    if (idx >= n) return;
    const int d = idx & 511;
    const int row = idx >> 9;
    const int l = row % L;
    const int rowbase = row - l;

    const float* ws[3] = { w0, w1, w2 };
    const float* bs[3] = { b0, b1, b2 };
    float best = -3.4e38f;
#pragma unroll
    for (int j = 0; j < 3; ++j) {
        const float* w = ws[j] + (size_t)d * k;
        float s = 0.0f;
        for (int t = 0; t < k; ++t) {
            int p = l + (t - (k - 1)) * dil;
            if (p >= 0) s += si[(size_t)(rowbase + p) * DMODEL + d] * w[t];
        }
        s += bs[j][d];
        best = fmaxf(best, s);
    }
    out[(size_t)row * DMODEL + d] = best;
}

__global__ __launch_bounds__(256)
void softmax128_kernel(float* __restrict__ s)
{
    const int row  = blockIdx.x * 8 + (threadIdx.x >> 5);
    const int lane = threadIdx.x & 31;
    float v[4];
#pragma unroll
    for (int e = 0; e < 4; ++e) v[e] = s[(size_t)row * 128 + lane + e * 32];
    float mx = fmaxf(fmaxf(v[0], v[1]), fmaxf(v[2], v[3]));
    for (int off = 16; off > 0; off >>= 1) mx = fmaxf(mx, __shfl_xor(mx, off, 32));
    float sum = 0.0f;
#pragma unroll
    for (int e = 0; e < 4; ++e) { v[e] = expf(v[e] - mx); sum += v[e]; }
    for (int off = 16; off > 0; off >>= 1) sum += __shfl_xor(sum, off, 32);
    const float inv = 1.0f / sum;
#pragma unroll
    for (int e = 0; e < 4; ++e) s[(size_t)row * 128 + lane + e * 32] = v[e] * inv;
}

__global__ void add2_kernel(const float* __restrict__ a, const float* __restrict__ b,
                            float* __restrict__ o, int n)
{
    int i = blockIdx.x * blockDim.x + threadIdx.x;
    if (i < n) o[i] = a[i] + b[i];
}

__global__ void combine3_kernel(const float* __restrict__ conn,
                                const float* __restrict__ pred,
                                const float* __restrict__ si,
                                float* __restrict__ o, int ldo, int coloff, int n)
{
    int i = blockIdx.x * blockDim.x + threadIdx.x;
    if (i >= n) return;
    int row = i >> 9, d = i & 511;
    o[(size_t)row * ldo + coloff + d] = conn[i] + 0.1f * pred[i] + si[i];
}

__global__ void copy512_kernel(const float* __restrict__ src,
                               float* __restrict__ dst, int ldd, int coloff, int n)
{
    int i = blockIdx.x * blockDim.x + threadIdx.x;
    if (i >= n) return;
    int row = i >> 9, d = i & 511;
    dst[(size_t)row * ldd + coloff + d] = src[i];
}

// ---------------------------------------------------------------------------
static inline void launch_gemm(const float* A, int lda, const float* B, int ldb,
                               int bTrans, const float* bias, float* C, int ldcv,
                               int ccol, int M, int N, int K, int gelu,
                               hipStream_t s)
{
    dim3 grid(N / BN, M / BM), blk(256);
    if (gelu)
        gemm_wmma_bf16<1, 0><<<grid, blk, 0, s>>>(A, lda, B, ldb, bTrans, bias,
                                                  C, ldcv, ccol, M, N, K);
    else
        gemm_wmma_bf16<0, 0><<<grid, blk, 0, s>>>(A, lda, B, ldb, bTrans, bias,
                                                  C, ldcv, ccol, M, N, K);
}

extern "C" void kernel_launch(void* const* d_in, const int* in_sizes, int n_in,
                              void* d_out, int out_size, void* d_ws, size_t ws_size,
                              hipStream_t stream)
{
    (void)in_sizes; (void)n_in; (void)out_size; (void)ws_size;

    const int M = MTOK, L = 2048, D = DMODEL;

    const int*   tokens     = (const int*)d_in[0];
    const float* emb        = (const float*)d_in[1];
    const float* pos_phases = (const float*)d_in[2];
    auto P = [&](int i, int j) { return (const float*)d_in[3 + i * 19 + j]; };
    const float* out_w1 = (const float*)d_in[60];
    const float* out_b1 = (const float*)d_in[61];
    const float* out_w2 = (const float*)d_in[62];
    const float* out_b2 = (const float*)d_in[63];

    // workspace carve-up
    float* ws        = (float*)d_ws;
    float* xemb      = ws;                                  // M*512 f32
    float* si        = xemb      + (size_t)M * D;
    float* conv      = si        + (size_t)M * D;
    float* scores    = conv      + (size_t)M * D;           // M*128
    float* combined  = scores    + (size_t)M * 128;         // M*1024
    float* mixout    = combined  + (size_t)M * 1024;
    float* connected = mixout    + (size_t)M * D;
    float* pred      = connected + (size_t)M * D;
    float* hbuf      = pred      + (size_t)M * D;           // M*2048 f32
    float* outs      = hbuf      + (size_t)M * 2048;        // M*1536 f32
    unsigned short* w2bf = (unsigned short*)(outs + (size_t)M * 1536); // 32000*2048 bf16
    unsigned short* hbf  = w2bf + (size_t)32000 * 2048;                // M*2048 bf16

    const int nElt = M * D;
    const dim3 eb(256), eg((nElt + 255) / 256);

    embed_kernel<<<eg, eb, 0, stream>>>(tokens, emb, xemb, nElt);

    const float* cur = xemb;
    int lda_cur = D;

    for (int i = 0; i < 3; ++i) {
        posln_kernel<<<M, 256, 0, stream>>>(cur, lda_cur, pos_phases + i * D,
                                            P(i, 0), P(i, 1), si, L);

        const int k   = (i == 0) ? 32 : (i == 1 ? 64 : 128);
        const int dil = i + 1;
        conv_kernel<<<eg, eb, 0, stream>>>(si, P(i, 2), P(i, 3), P(i, 4),
                                           P(i, 5), P(i, 6), P(i, 7),
                                           conv, L, k, dil, nElt);

        launch_gemm(conv, D, P(i, 9), D, 0, P(i, 10), scores, 128, 0,
                    M, 128, D, 0, stream);
        softmax128_kernel<<<M / 8, 256, 0, stream>>>(scores);

        launch_gemm(scores, 128, P(i, 8), D, 1, nullptr, combined, 1024, 512,
                    M, D, 128, 0, stream);
        copy512_kernel<<<eg, eb, 0, stream>>>(conv, combined, 1024, 0, nElt);

        launch_gemm(combined, 1024, P(i, 11), 1024, 0, P(i, 12), mixout, D, 0,
                    M, D, 1024, 0, stream);
        add2_kernel<<<eg, eb, 0, stream>>>(mixout, conv, connected, nElt);

        launch_gemm(connected, D, P(i, 13), D, 0, P(i, 14), hbuf, 2048, 0,
                    M, 2048, D, 1, stream);
        launch_gemm(hbuf, 2048, P(i, 15), 2048, 0, P(i, 16), pred, D, 0,
                    M, D, 2048, 0, stream);

        if (i == 0) {
            combine3_kernel<<<eg, eb, 0, stream>>>(connected, pred, si,
                                                   outs, 1536, 0, nElt);
        } else {
            combine3_kernel<<<eg, eb, 0, stream>>>(connected, pred, si,
                                                   mixout, D, 0, nElt);
            copy512_kernel<<<eg, eb, 0, stream>>>(mixout, outs, 1536, i * D, nElt);
            launch_gemm(outs, 1536, P(i, 17), D * (i + 1), 0, P(i, 18),
                        conv, D, 0, M, D, D * (i + 1), 0, stream);
            copy512_kernel<<<eg, eb, 0, stream>>>(conv, outs, 1536, i * D, nElt);
        }
        cur = outs + (size_t)i * D;
        lda_cur = 1536;
    }

    // ---- output head ----
    // hidden = GELU(cur @ out_w1.T + b1), written directly as bf16
    {
        dim3 grid(2048 / BN, M / BM), blk(256);
        gemm_wmma_bf16<1, 1><<<grid, blk, 0, stream>>>(
            cur, 1536, out_w1, D, 0, out_b1, (void*)hbf, 2048, 0, M, 2048, D);
    }
    // out_w2 -> bf16 once per call (393 MB traffic ~ 17us @ 23.3 TB/s)
    {
        const long n4 = (long)32000 * 2048 / 4;
        f32_to_bf16_kernel<<<(unsigned)(n4 / 256), 256, 0, stream>>>(
            out_w2, w2bf, n4);
    }
    // logits = hidden @ out_w2.T + b2 : bf16 operands, async LDS staging
    {
        dim3 grid(32000 / BN, M / BM), blk(256);
        gemm_wmma_bf16_async<<<grid, blk, 0, stream>>>(
            hbf, 2048, w2bf, 2048, out_b2, (float*)d_out, 32000, M, 32000, 2048);
    }
}
